// TemporalGNN_57887569215851
// MI455X (gfx1250) — matrix operand order repaired and checked by reading, not compiled, not run on hardware
//
#include <hip/hip_runtime.h>
#include <hip/hip_bf16.h>
#include <math.h>
#include <stdint.h>

typedef __attribute__((ext_vector_type(16))) _Float16 v16h;
typedef __attribute__((ext_vector_type(8)))  _Float16 v8h;
typedef __attribute__((ext_vector_type(8)))  float    v8f;
typedef __attribute__((ext_vector_type(4)))  float    v4f;

#define NN     1000
#define NFEAT  128
#define NHID   128
#define BB     8
#define TT     20
#define NSNAP  (BB*TT)
#define NIDX   100

#define KMAIN  992            // 31 full 32-wide K steps (k < 992 <= NN-8)
#define KTAIL  992            // tail block start; valid k: 992..999 (8 cols)

// ---------------------------------------------------------------------------
// CDNA5 async global->LDS copy (tracked by ASYNCcnt).
// lds_addr: 32-bit LDS byte address (low 32 bits of generic shared pointer).
// ---------------------------------------------------------------------------
__device__ __forceinline__ void async_copy_b128(uint32_t lds_addr, const void* gptr) {
    asm volatile("global_load_async_to_lds_b128 %0, %1, off"
                 :: "v"(lds_addr), "v"((uint64_t)(uintptr_t)gptr)
                 : "memory");
}
__device__ __forceinline__ void wait_asynccnt0() {
    asm volatile("s_wait_asynccnt 0x0" ::: "memory");
}

// ---------------------------------------------------------------------------
// fp32 -> f16 elementwise convert
// ---------------------------------------------------------------------------
__global__ void cvt_f32_to_f16_kernel(const float* __restrict__ src,
                                      _Float16* __restrict__ dst, size_t n) {
    size_t i = (size_t)blockIdx.x * blockDim.x + threadIdx.x;
    if (i < n) dst[i] = (_Float16)src[i];
}

// ---------------------------------------------------------------------------
// Shared WMMA tile compute: 8 waves, WG tile 128 rows x 64 cols.
// Wave w: rows [(w>>1)*32, +32), cols [(w&1)*32, +32) -> 4 accumulators.
// As: [row][k]  (rows 80B, 16B-aligned)   Bs: [n][k] transposed (rows 96B)
// ---------------------------------------------------------------------------
struct Accs { v8f c00, c01, c10, c11; };

__device__ __forceinline__
void tile_mma(const _Float16 (*As)[40], const _Float16 (*Bs)[48],
              int lane, int wm, int wn, Accs& a)
{
    const int mr0 = wm * 32 + (lane & 15);
    const int mr1 = mr0 + 16;
    const int nr0 = wn * 32 + (lane & 15);
    const int nr1 = nr0 + 16;
    const int ka  = (lane >= 16) ? 8  : 0;   // A frag: hi lanes hold K+8
    const int kb  = (lane >= 16) ? 8  : 0;   // B frag: hi lanes hold K+8
    union { v16h v; v8h h[2]; } a0, a1;
    a0.h[0] = *(const v8h*)&As[mr0][ka];
    a0.h[1] = *(const v8h*)&As[mr0][ka + 16];
    a1.h[0] = *(const v8h*)&As[mr1][ka];
    a1.h[1] = *(const v8h*)&As[mr1][ka + 16];
    union { v16h v; v8h h[2]; } bb0, bb1;
    bb0.h[0] = *(const v8h*)&Bs[nr0][kb];
    bb0.h[1] = *(const v8h*)&Bs[nr0][kb + 16];
    bb1.h[0] = *(const v8h*)&Bs[nr1][kb];
    bb1.h[1] = *(const v8h*)&Bs[nr1][kb + 16];
    v16h b0 = bb0.v;
    v16h b1 = bb1.v;
    a.c00 = __builtin_amdgcn_wmma_f32_16x16x32_f16(false, a0.v, false, b0, (short)0, a.c00, false, false);
    a.c01 = __builtin_amdgcn_wmma_f32_16x16x32_f16(false, a0.v, false, b1, (short)0, a.c01, false, false);
    a.c10 = __builtin_amdgcn_wmma_f32_16x16x32_f16(false, a1.v, false, b0, (short)0, a.c10, false, false);
    a.c11 = __builtin_amdgcn_wmma_f32_16x16x32_f16(false, a1.v, false, b1, (short)0, a.c11, false, false);
}

// ---------------------------------------------------------------------------
// Aggregation GEMM:  C[1000 x Nc] = adj[1000 x 1000](f32) * B[1000 x Nc](f16)
// Nc in {64,128}. Workgroup tile 128x64.
// ---------------------------------------------------------------------------
__global__ __launch_bounds__(256)
void agg_gemm_kernel(const float* __restrict__ adjbase,
                     const _Float16* __restrict__ Bbase,
                     _Float16* __restrict__ Cbase, int Nc)
{
    __shared__ _Float16 As[128][40];
    __shared__ _Float16 Bs[64][48];

    const int tid  = threadIdx.x;
    const int lane = tid & 31;
    const int wave = tid >> 5;
    const int wn   = wave & 1;
    const int wm   = wave >> 1;
    const int snap = blockIdx.z;
    const int m0   = blockIdx.x * 128;
    const int n0   = blockIdx.y * 64;

    const float*    A  = adjbase + (size_t)snap * NN * NN;
    const _Float16* Bp = Bbase   + (size_t)snap * NN * Nc;
    _Float16*       Cp = Cbase   + (size_t)snap * NN * Nc;

    Accs acc; acc.c00 = (v8f){}; acc.c01 = (v8f){}; acc.c10 = (v8f){}; acc.c11 = (v8f){};

    const int rowA = tid >> 1;            // 0..127
    const int colA = (tid & 1) * 16;      // 0 / 16
    const int kB   = tid >> 3;            // 0..31
    const int nB   = (tid & 7) * 8;       // 0..56

    const int  mA     = m0 + rowA;
    const int  mAc    = (mA < NN) ? mA : (NN - 1);      // clamped row
    const bool mok    = (mA < NN);
    const float* srcA = A + (size_t)mAc * NN + colA;    // + k0 per iter

    // ---- main loop: 31 iterations, no k guards anywhere
    for (int k0 = 0; k0 < KMAIN; k0 += 32) {
        __syncthreads();
        {
            const float* s = srcA + k0;
            __builtin_prefetch(s + 32, 0, 0);           // next K tile of adj
            #pragma unroll
            for (int i = 0; i < 4; ++i) {
                v4f f = *(const v4f*)(s + i * 4);
                As[rowA][colA + i*4 + 0] = (_Float16)(mok ? f.x : 0.0f);
                As[rowA][colA + i*4 + 1] = (_Float16)(mok ? f.y : 0.0f);
                As[rowA][colA + i*4 + 2] = (_Float16)(mok ? f.z : 0.0f);
                As[rowA][colA + i*4 + 3] = (_Float16)(mok ? f.w : 0.0f);
            }
        }
        {
            const _Float16* s = Bp + (size_t)(k0 + kB) * Nc + (n0 + nB);
            #pragma unroll
            for (int i = 0; i < 8; ++i) Bs[nB + i][kB] = s[i];
        }
        __syncthreads();
        tile_mma(As, Bs, lane, wm, wn, acc);
    }

    // ---- tail: k = 992..999 valid (8 cols), branch-free clamped loads
    __syncthreads();
    {
        #pragma unroll
        for (int i = 0; i < 16; ++i) {
            const int k  = KTAIL + colA + i;
            const int kc = (k < NN) ? k : (NN - 1);
            const float f = A[(size_t)mAc * NN + kc];
            As[rowA][colA + i] = (_Float16)((mok && k < NN) ? f : 0.0f);
        }
        const int  k  = KTAIL + kB;
        const int  kc = (k < NN) ? k : (NN - 1);
        const bool kok = (k < NN);
        const _Float16* s = Bp + (size_t)kc * Nc + (n0 + nB);
        #pragma unroll
        for (int i = 0; i < 8; ++i) {
            const _Float16 v = s[i];
            Bs[nB + i][kB] = kok ? v : (_Float16)0.0f;
        }
    }
    __syncthreads();
    tile_mma(As, Bs, lane, wm, wn, acc);

    // ---- store (C layout: VGPR r -> M = r (+8 hi lanes), N = lane&15)
    const int nc0 = n0 + wn * 32 + (lane & 15);
    const int nc1 = nc0 + 16;
    const int mb  = m0 + wm * 32 + ((lane >= 16) ? 8 : 0);
    #pragma unroll
    for (int r = 0; r < 8; ++r) {
        const int m1 = mb + r;
        const int m2 = m1 + 16;
        if (m1 < NN) {
            Cp[(size_t)m1 * Nc + nc0] = (_Float16)acc.c00[r];
            Cp[(size_t)m1 * Nc + nc1] = (_Float16)acc.c01[r];
        }
        if (m2 < NN) {
            Cp[(size_t)m2 * Nc + nc0] = (_Float16)acc.c10[r];
            Cp[(size_t)m2 * Nc + nc1] = (_Float16)acc.c11[r];
        }
    }
}

// ---------------------------------------------------------------------------
// Layer GEMM with up to 3 concatenated f16 A-segments (widths multiple of 32),
// fp32 weights W[Ktot x Nc], fused bias + ReLU, f16 output.
// Out = relu( [A0|A1|A2] @ W + bias ).  Ktot multiple of 32 -> no K tail.
// A tile staged with CDNA5 async global->LDS DMA (ASYNCcnt).
// ---------------------------------------------------------------------------
__global__ __launch_bounds__(256)
void layer_gemm_kernel(const _Float16* __restrict__ A0,
                       const _Float16* __restrict__ A1,
                       const _Float16* __restrict__ A2,
                       int w0, int w1, int w2,
                       const float* __restrict__ W,
                       const float* __restrict__ bias,
                       _Float16* __restrict__ OutBase,
                       int Ktot, int Nc)
{
    __shared__ _Float16 As[128][40];
    __shared__ _Float16 Bs[64][48];

    const int tid  = threadIdx.x;
    const int lane = tid & 31;
    const int wave = tid >> 5;
    const int wn   = wave & 1;
    const int wm   = wave >> 1;
    const int snap = blockIdx.z;
    const int m0   = blockIdx.x * 128;
    const int n0   = blockIdx.y * 64;

    _Float16* Out = OutBase + (size_t)snap * NN * Nc;

    Accs acc; acc.c00 = (v8f){}; acc.c01 = (v8f){}; acc.c10 = (v8f){}; acc.c11 = (v8f){};

    const int rowA = tid >> 1;
    const int colA = (tid & 1) * 16;
    const int kB   = tid >> 3;
    const int nB   = (tid & 7) * 8;

    const int  mA  = m0 + rowA;
    const bool mok = (mA < NN);
    const bool nokB = (n0 + nB) < Nc;       // uniform per thread (nB mult of 8, Nc mult of 32)
    const uint32_t ldsA = (uint32_t)(uintptr_t)&As[rowA][colA];   // LDS byte address

    for (int k0 = 0; k0 < Ktot; k0 += 32) {
        // segment select (32-block always inside one segment)
        const _Float16* Ap; int ldA; int kbase;
        if (k0 < w0)            { Ap = A0 + (size_t)snap * NN * w0; ldA = w0; kbase = 0;       }
        else if (k0 < w0 + w1)  { Ap = A1 + (size_t)snap * NN * w1; ldA = w1; kbase = w0;      }
        else                    { Ap = A2 + (size_t)snap * NN * w2; ldA = w2; kbase = w0 + w1; }

        __syncthreads();
        // ---- A tile: async global->LDS DMA, 2 x 16B per thread
        if (mok) {
            const _Float16* s = Ap + (size_t)mA * ldA + (k0 - kbase) + colA;
            async_copy_b128(ldsA,      s);
            async_copy_b128(ldsA + 16, s + 8);
        } else {
            *(v8h*)&As[rowA][colA]     = (v8h){};
            *(v8h*)&As[rowA][colA + 8] = (v8h){};
        }
        // ---- weights: transpose + fp32->f16 through VGPRs (overlaps async DMA)
        {
            if (nokB) {
                const float* s = W + (size_t)(k0 + kB) * Nc + (n0 + nB);
                #pragma unroll
                for (int i = 0; i < 8; ++i) Bs[nB + i][kB] = (_Float16)s[i];
            } else {
                #pragma unroll
                for (int i = 0; i < 8; ++i) Bs[nB + i][kB] = (_Float16)0.0f;
            }
        }
        wait_asynccnt0();            // own async copies landed in LDS
        __syncthreads();             // cross-wave visibility
        tile_mma(As, Bs, lane, wm, wn, acc);
    }

    // epilogue: bias + ReLU, f16 store, col-guarded for Nc=32
    const int nc0 = n0 + wn * 32 + (lane & 15);
    const int nc1 = nc0 + 16;
    const int mb  = m0 + wm * 32 + ((lane >= 16) ? 8 : 0);
    const bool n0ok = (nc0 < Nc);
    const bool n1ok = (nc1 < Nc);
    const float bv0 = bias[n0ok ? nc0 : 0];
    const float bv1 = bias[n1ok ? nc1 : 0];
    #pragma unroll
    for (int r = 0; r < 8; ++r) {
        const int m1 = mb + r;
        const int m2 = m1 + 16;
        if (m1 < NN) {
            if (n0ok) Out[(size_t)m1 * Nc + nc0] = (_Float16)fmaxf(acc.c00[r] + bv0, 0.0f);
            if (n1ok) Out[(size_t)m1 * Nc + nc1] = (_Float16)fmaxf(acc.c01[r] + bv1, 0.0f);
        }
        if (m2 < NN) {
            if (n0ok) Out[(size_t)m2 * Nc + nc0] = (_Float16)fmaxf(acc.c10[r] + bv0, 0.0f);
            if (n1ok) Out[(size_t)m2 * Nc + nc1] = (_Float16)fmaxf(acc.c11[r] + bv1, 0.0f);
        }
    }
}

// ---------------------------------------------------------------------------
// Readout: g[snap, f] = mean_i H[snap, idx[i], f]
// ---------------------------------------------------------------------------
__global__ void readout_kernel(const _Float16* __restrict__ H,
                               const int* __restrict__ idx,
                               float* __restrict__ g)
{
    const int snap = blockIdx.x;
    const int f    = threadIdx.x;        // 128
    float s = 0.0f;
    for (int i = 0; i < NIDX; ++i) {
        const int n = idx[i];
        s += (float)H[((size_t)snap * NN + n) * NHID + f];
    }
    g[(size_t)snap * NHID + f] = s * (1.0f / NIDX);
}

// ---------------------------------------------------------------------------
// xW[snap, u] = g[snap,:] @ W_ih[u,:] + b_ih[u] + b_hh[u]     (u in [0,512))
// ---------------------------------------------------------------------------
__global__ void xw_kernel(const float* __restrict__ g,
                          const float* __restrict__ W_ih,
                          const float* __restrict__ b_ih,
                          const float* __restrict__ b_hh,
                          float* __restrict__ xW)
{
    __shared__ float gs[NHID];
    const int snap = blockIdx.x;
    const int tid  = threadIdx.x;        // 256
    if (tid < NHID) gs[tid] = g[(size_t)snap * NHID + tid];
    __syncthreads();
    #pragma unroll
    for (int r = 0; r < 2; ++r) {
        const int u = tid * 2 + r;
        float s = b_ih[u] + b_hh[u];
        for (int k = 0; k < NHID; ++k) s += W_ih[(size_t)u * NHID + k] * gs[k];
        xW[(size_t)snap * 4 * NHID + u] = s;
    }
}

// ---------------------------------------------------------------------------
// Sequential LSTM over T=20 steps + gather at lengths-1 + final head.
// ---------------------------------------------------------------------------
__global__ __launch_bounds__(512)
void lstm_kernel(const float* __restrict__ xW,
                 const float* __restrict__ W_hh,
                 const int* __restrict__ lengths,
                 const float* __restrict__ Wfc,
                 const float* __restrict__ bfc,
                 float* __restrict__ out)
{
    __shared__ float hs[BB][NHID];
    __shared__ float cs[BB][NHID];
    __shared__ float gates[BB][4 * NHID];
    __shared__ float last[BB][NHID];

    const int tid = threadIdx.x;
    for (int i = tid; i < BB * NHID; i += 512) {
        ((float*)hs)[i] = 0.0f; ((float*)cs)[i] = 0.0f; ((float*)last)[i] = 0.0f;
    }
    __syncthreads();

    for (int t = 0; t < TT; ++t) {
        float acc[BB];
        #pragma unroll
        for (int b = 0; b < BB; ++b)
            acc[b] = xW[(size_t)(b * TT + t) * 4 * NHID + tid];
        for (int k = 0; k < NHID; ++k) {
            const float w = W_hh[(size_t)tid * NHID + k];
            #pragma unroll
            for (int b = 0; b < BB; ++b) acc[b] += w * hs[b][k];
        }
        #pragma unroll
        for (int b = 0; b < BB; ++b) gates[b][tid] = acc[b];
        __syncthreads();
        #pragma unroll
        for (int r = 0; r < 2; ++r) {
            const int cell = tid * 2 + r;
            const int b = cell >> 7, j = cell & (NHID - 1);
            const float ig = 1.0f / (1.0f + __expf(-gates[b][j]));
            const float fg = 1.0f / (1.0f + __expf(-gates[b][NHID + j]));
            const float gg = tanhf(gates[b][2 * NHID + j]);
            const float og = 1.0f / (1.0f + __expf(-gates[b][3 * NHID + j]));
            const float c  = fg * cs[b][j] + ig * gg;
            const float h  = og * tanhf(c);
            cs[b][j] = c; hs[b][j] = h;
            if (t == lengths[b] - 1) last[b][j] = h;
        }
        __syncthreads();
    }
    if (tid < BB) {
        float s = bfc[0];
        for (int k = 0; k < NHID; ++k) s += last[tid][k] * Wfc[k];
        out[tid] = s;
    }
}

// ---------------------------------------------------------------------------
extern "C" void kernel_launch(void* const* d_in, const int* in_sizes, int n_in,
                              void* d_out, int out_size, void* d_ws, size_t ws_size,
                              hipStream_t stream)
{
    const float* adj  = (const float*)d_in[0];
    const float* feat = (const float*)d_in[1];
    const float* Wg1  = (const float*)d_in[2];  const float* bg1 = (const float*)d_in[3];
    const float* Wg2  = (const float*)d_in[4];  const float* bg2 = (const float*)d_in[5];
    const float* Wg3  = (const float*)d_in[6];  const float* bg3 = (const float*)d_in[7];
    const float* Wgo  = (const float*)d_in[8];  const float* bgo = (const float*)d_in[9];
    const float* W_ih = (const float*)d_in[10]; const float* W_hh = (const float*)d_in[11];
    const float* b_ih = (const float*)d_in[12]; const float* b_hh = (const float*)d_in[13];
    const float* Wfc  = (const float*)d_in[14]; const float* bfc  = (const float*)d_in[15];
    const int*   idx  = (const int*)d_in[16];   const int* lengths = (const int*)d_in[17];
    float* out = (float*)d_out;

    char* ws = (char*)d_ws;
    size_t off = 0;
    auto carve = [&](size_t bytes) -> void* {
        void* p = (void*)(ws + off);
        off += (bytes + 255) & ~(size_t)255;
        return p;
    };
    const size_t SN = NSNAP;
    _Float16* Xh  = (_Float16*)carve(SN * NN * NFEAT * sizeof(_Float16));
    _Float16* AX  = (_Float16*)carve(SN * NN * 128   * sizeof(_Float16));
    _Float16* H1  = (_Float16*)carve(SN * NN * 128   * sizeof(_Float16));
    _Float16* AH1 = (_Float16*)carve(SN * NN * 128   * sizeof(_Float16));
    _Float16* H2  = (_Float16*)carve(SN * NN * 64    * sizeof(_Float16));
    _Float16* AH2 = (_Float16*)carve(SN * NN * 64    * sizeof(_Float16));
    _Float16* H3  = (_Float16*)carve(SN * NN * 32    * sizeof(_Float16));
    _Float16* Hf  = (_Float16*)carve(SN * NN * 128   * sizeof(_Float16));
    float*    g   = (float*)   carve(SN * NHID       * sizeof(float));
    float*    xWb = (float*)   carve(SN * 4 * NHID   * sizeof(float));

    {
        size_t n = SN * NN * (size_t)NFEAT;
        cvt_f32_to_f16_kernel<<<dim3((unsigned)((n + 255) / 256)), 256, 0, stream>>>(feat, Xh, n);
    }
    // 1) AX = adj @ X
    agg_gemm_kernel<<<dim3(8, 2, NSNAP), 256, 0, stream>>>(adj, Xh, AX, 128);
    // 2) H1 = relu(AX @ Wg1 + bg1)
    layer_gemm_kernel<<<dim3(8, 2, NSNAP), 256, 0, stream>>>(AX, AX, AX, 128, 0, 0,
                                                             Wg1, bg1, H1, 128, 128);
    // 3) AH1 = adj @ H1
    agg_gemm_kernel<<<dim3(8, 2, NSNAP), 256, 0, stream>>>(adj, H1, AH1, 128);
    // 4) H2 = relu([AH1 | AX] @ Wg2 + bg2)   (agg of concat == concat of aggs)
    layer_gemm_kernel<<<dim3(8, 1, NSNAP), 256, 0, stream>>>(AH1, AX, AX, 128, 128, 0,
                                                             Wg2, bg2, H2, 256, 64);
    // 5) AH2 = adj @ H2
    agg_gemm_kernel<<<dim3(8, 1, NSNAP), 256, 0, stream>>>(adj, H2, AH2, 64);
    // 6) H3 = relu([AH2 | AH1] @ Wg3 + bg3)
    layer_gemm_kernel<<<dim3(8, 1, NSNAP), 256, 0, stream>>>(AH2, AH1, AH1, 64, 128, 0,
                                                             Wg3, bg3, H3, 192, 32);
    // 7) H = relu([H3 | H2 | H1] @ Wgo + bgo)
    layer_gemm_kernel<<<dim3(8, 2, NSNAP), 256, 0, stream>>>(H3, H2, H1, 32, 64, 128,
                                                             Wgo, bgo, Hf, 224, 128);
    // 8) readout mean over idx
    readout_kernel<<<dim3(NSNAP), 128, 0, stream>>>(Hf, idx, g);
    // 9) input-to-hidden gate precompute
    xw_kernel<<<dim3(NSNAP), 256, 0, stream>>>(g, W_ih, b_ih, b_hh, xWb);
    // 10) sequential LSTM + gather + head
    lstm_kernel<<<dim3(1), 512, 0, stream>>>(xWb, W_hh, lengths, Wfc, bfc, out);
}
